// AttentionBlock_10892037063162
// MI455X (gfx1250) — matrix operand order repaired and checked by reading, not compiled, not run on hardware
//
#include <hip/hip_runtime.h>
#include <hip/hip_bf16.h>
#include <math.h>

// ---------------------------------------------------------------------------
// AttentionBlock for MI455X (gfx1250, wave32, WMMA f32_16x16x32_f16)
// B=16, C=512, T=1024, heads=8, hd=64
// ---------------------------------------------------------------------------

typedef __attribute__((ext_vector_type(16))) _Float16 v16h;
typedef __attribute__((ext_vector_type(8)))  _Float16 v8h;
typedef __attribute__((ext_vector_type(8)))  float    v8f;
typedef __attribute__((ext_vector_type(4)))  int      v4i;

#define BB     16
#define CC     512
#define TT     1024
#define HEADS  8
#define HD     64
#define OC3    1536
#define EPSF   1e-5f
#define SCALE_QK 0.35355339059327373f   /* 1/64^(1/4) */
#define L2E    1.44269504088896340736f

// workspace layout (bytes)
#define OFF_WQ  ((size_t)0)
#define OFF_WP  (OFF_WQ + (size_t)OC3*CC*2)          // 1.5 MB
#define OFF_XN  (OFF_WP + (size_t)CC*CC*2)           // xn[b][t][c] f16
#define OFF_Q   (OFF_XN + (size_t)BB*TT*CC*2)        // q [bh][t][hd] f16 (pre-scaled)
#define OFF_K   (OFF_Q  + (size_t)BB*CC*TT*2)        // k [bh][s][hd] f16 (pre-scaled)
#define OFF_V   (OFF_K  + (size_t)BB*CC*TT*2)        // v [bh][hd][s] f16
#define OFF_HT  (OFF_V  + (size_t)BB*CC*TT*2)        // h [b][t][c]  f16

// ---- async Global -> LDS staging (CDNA5 GLOBAL_LOAD_ASYNC_TO_LDS_B128) ----
#if __has_builtin(__builtin_amdgcn_global_load_async_to_lds_b128)
#define HAVE_ASYNC_LDS 1
typedef __attribute__((address_space(1))) v4i* gas_i4p;
typedef __attribute__((address_space(3))) v4i* las_i4p;
#endif

static __device__ __forceinline__ void stage16B(const _Float16* g, _Float16* l) {
#ifdef HAVE_ASYNC_LDS
    __builtin_amdgcn_global_load_async_to_lds_b128((gas_i4p)g, (las_i4p)l, 0, 0);
#else
    *(v8h*)l = *(const v8h*)g;
#endif
}

static __device__ __forceinline__ void wait_async() {
#ifdef HAVE_ASYNC_LDS
#if __has_builtin(__builtin_amdgcn_s_wait_asynccnt)
    __builtin_amdgcn_s_wait_asynccnt(0);
#else
    asm volatile("s_wait_asynccnt 0x0" ::: "memory");
#endif
#endif
}

static __device__ __forceinline__ v8f wmma_f16(v16h a, v16h b, v8f c) {
    return __builtin_amdgcn_wmma_f32_16x16x32_f16(false, a, false, b,
                                                  (short)0, c, false, false);
}

// A fragment 16x32 f16 from row-major [ld] memory (row=M, K contiguous).
// ISA layout: lane l: M=l&15; lane<16 holds K={0..7,16..23}, lane>=16 K={8..15,24..31}
static __device__ __forceinline__ v16h load_frag_a(const _Float16* base, int ld) {
    const int l  = threadIdx.x & 31;
    const int m  = l & 15;
    const int kb = (l >> 4) * 8;
    const _Float16* p = base + (size_t)m * ld + kb;
    union { v16h v; v8h h[2]; } u;
    u.h[0] = *(const v8h*)(p);        // K = kb .. kb+7
    u.h[1] = *(const v8h*)(p + 16);   // K = kb+16 .. kb+23
    return u.v;
}

// B fragment 32x16 f16; memory holds column n at base+n*ld with K contiguous.
// ISA layout: lane l: N=l&15; lane<16 holds K={0..7,16..23}, lane>=16 K={8..15,24..31}
static __device__ __forceinline__ v16h load_frag_b(const _Float16* base, int ld) {
    const int l  = threadIdx.x & 31;
    const int n  = l & 15;
    const int ks = (l >> 4) * 8;
    const _Float16* p = base + (size_t)n * ld + ks;
    union { v16h v; v8h h[2]; } u;
    u.h[0] = *(const v8h*)(p);
    u.h[1] = *(const v8h*)(p + 16);
    return u.v;
}

// ---------------------------------------------------------------------------
// K0: convert w_qkv / w_proj to f16
// ---------------------------------------------------------------------------
__global__ void k_prep_weights(const float* __restrict__ w_qkv,
                               const float* __restrict__ w_proj,
                               char* __restrict__ ws) {
    _Float16* wq = (_Float16*)(ws + OFF_WQ);
    _Float16* wp = (_Float16*)(ws + OFF_WP);
    const int nq = OC3 * CC;              // 786432
    int idx = blockIdx.x * blockDim.x + threadIdx.x;
    if (idx < nq)            wq[idx]      = (_Float16)w_qkv[idx];
    else                     wp[idx - nq] = (_Float16)w_proj[idx - nq];
}

// ---------------------------------------------------------------------------
// K1: LayerNorm over C for each (b,t); write xn transposed [b][t][c] f16.
// Two passes over x; x (33 MB) is L2-resident (192 MB L2).
// ---------------------------------------------------------------------------
__global__ void k_layernorm(const float* __restrict__ x, char* __restrict__ ws) {
    _Float16* xn = (_Float16*)(ws + OFF_XN);
    const int b  = blockIdx.x >> 4;            // 16 blocks per batch
    const int t0 = (blockIdx.x & 15) * 64;
    const int t  = t0 + threadIdx.x;           // 64 threads: one t each

    const float* xb = x + (size_t)b * CC * TT + t;
    float s = 0.f, ss = 0.f;
    for (int c = 0; c < CC; ++c) {
        float v = xb[(size_t)c * TT];
        s += v; ss += v * v;
    }
    const float mu   = s * (1.0f / CC);
    const float var  = ss * (1.0f / CC) - mu * mu;
    const float rstd = rsqrtf(var + EPSF);

    _Float16* o = xn + ((size_t)b * TT + t) * CC;
    for (int c = 0; c < CC; ++c) {
        float v = xb[(size_t)c * TT];
        o[c] = (_Float16)((v - mu) * rstd);
    }
}

// ---------------------------------------------------------------------------
// K2: QKV GEMM.  O[b, o, t] = sum_c Wqkv[o,c] * xn[b,t,c] + bias[o].
// Each wave computes a 16(M=o) x 64(N=t) tile, K-loop over C in steps of 32.
// ---------------------------------------------------------------------------
__global__ void k_qkv_gemm(const float* __restrict__ b_qkv, char* __restrict__ ws) {
    const _Float16* WQ = (const _Float16*)(ws + OFF_WQ);
    const _Float16* XN = (const _Float16*)(ws + OFF_XN);
    _Float16* Qp = (_Float16*)(ws + OFF_Q);
    _Float16* Kp = (_Float16*)(ws + OFF_K);
    _Float16* Vp = (_Float16*)(ws + OFF_V);

    const int gw = blockIdx.x * 8 + (threadIdx.x >> 5);   // global wave id
    const int tt = gw & 15;                               // 16 N-tiles of 64
    const int ot = (gw >> 4) % 96;                        // 96 M-tiles of 16
    const int b  = gw / (16 * 96);
    const int o0 = ot * 16;
    const int t0 = tt * 64;

    v8f acc[4];
    #pragma unroll
    for (int i = 0; i < 4; ++i) acc[i] = (v8f)(0.0f);

    for (int kc = 0; kc < CC; kc += 32) {
        v16h a = load_frag_a(WQ + (size_t)o0 * CC + kc, CC);
        #pragma unroll
        for (int nn = 0; nn < 4; ++nn) {
            v16h bf = load_frag_b(XN + ((size_t)b * TT + t0 + nn * 16) * CC + kc, CC);
            acc[nn] = wmma_f16(a, bf, acc[nn]);
        }
    }

    const int l  = threadIdx.x & 31;
    const int hi = l >> 4;
    const int n  = l & 15;
    #pragma unroll
    for (int nn = 0; nn < 4; ++nn) {
        #pragma unroll
        for (int j = 0; j < 8; ++j) {
            const int o = o0 + j + 8 * hi;
            const int t = t0 + nn * 16 + n;
            float val = acc[nn][j] + b_qkv[o];
            const int seg  = o >> 9;          // 0=q 1=k 2=v
            const int oc   = o & 511;
            const int head = oc >> 6;
            const int c    = oc & 63;
            const int bh   = b * HEADS + head;
            if (seg == 0)      Qp[((size_t)bh * TT + t) * HD + c] = (_Float16)(val * SCALE_QK);
            else if (seg == 1) Kp[((size_t)bh * TT + t) * HD + c] = (_Float16)(val * SCALE_QK);
            else               Vp[((size_t)bh * HD + c) * TT + t] = (_Float16)val;
        }
    }
}

// ---------------------------------------------------------------------------
// K3: flash attention per head. Block = 4 waves; wave owns 16 t-rows.
// K/V tiles (32 s-cols) are cooperatively staged into LDS per block via the
// CDNA5 async Global->LDS path, then consumed 4x by the waves as B fragments.
// ---------------------------------------------------------------------------
#define KPAD 72   /* 64 + 8 halves: 144B row stride, 16B aligned, conflict-free */
#define VPAD 40   /* 32 + 8 halves:  80B row stride, 16B aligned, conflict-free */

__global__ void k_attention(char* __restrict__ ws) {
    const _Float16* Qp = (const _Float16*)(ws + OFF_Q);
    const _Float16* Kp = (const _Float16*)(ws + OFF_K);
    const _Float16* Vp = (const _Float16*)(ws + OFF_V);
    _Float16* HT = (_Float16*)(ws + OFF_HT);

    __shared__ __align__(16) _Float16 Ks[32][KPAD];    // staged K: [s_local][c]
    __shared__ __align__(16) _Float16 Vs[64][VPAD];    // staged V: [c][s_local]
    __shared__ __align__(16) _Float16 Pb[4][16][32];   // per-wave P staging

    const int bh = blockIdx.x >> 4;
    const int tt = blockIdx.x & 15;
    const int w  = threadIdx.x >> 5;
    const int t0 = tt * 64 + w * 16;
    const int l  = threadIdx.x & 31;
    const int hi = l >> 4;
    const int n  = l & 15;
    const int ks = hi * 8;

    // resident Q fragments (rows t0..t0+15, K = channel 0..63)
    const _Float16* qbase = Qp + ((size_t)bh * TT + t0) * HD;
    const v16h qa0 = load_frag_a(qbase, HD);       // c 0..31
    const v16h qa1 = load_frag_a(qbase + 32, HD);  // c 32..63

    v8f acc[4];
    #pragma unroll
    for (int i = 0; i < 4; ++i) acc[i] = (v8f)(0.0f);
    float m8[8], l8[8];
    #pragma unroll
    for (int j = 0; j < 8; ++j) { m8[j] = -INFINITY; l8[j] = 0.0f; }

    const _Float16* Kg = Kp + (size_t)bh * TT * HD;   // [s][64]
    const _Float16* Vg = Vp + (size_t)bh * HD * TT;   // [c][1024]

    for (int s0 = 0; s0 < TT; s0 += 32) {
        if (s0 + 32 < TT) {
            __builtin_prefetch((const void*)(Kg + (size_t)(s0 + 32) * HD), 0, 1);
            __builtin_prefetch((const void*)(Vg + s0 + 32), 0, 1);
        }
        // ---- cooperative staging: 4KB K-tile + 4KB V-tile, 16B per op ----
        #pragma unroll
        for (int r = 0; r < 2; ++r) {
            const int hidx = ((int)threadIdx.x + r * 128) * 8;   // half index
            const int srow = hidx >> 6, scol = hidx & 63;        // K tile 32x64
            stage16B(Kg + (size_t)(s0 + srow) * HD + scol, &Ks[srow][scol]);
            const int vrow = hidx >> 5, vcol = hidx & 31;        // V tile 64x32
            stage16B(Vg + (size_t)vrow * TT + s0 + vcol, &Vs[vrow][vcol]);
        }
        wait_async();
        __syncthreads();

        // ---- scores for two 16-wide s tiles (K from LDS) ----
        v8f st[2];
        #pragma unroll
        for (int si = 0; si < 2; ++si) {
            v8f s = (v8f)(0.0f);
            #pragma unroll
            for (int cf = 0; cf < 2; ++cf) {          // c chunks 0..31 / 32..63
                union { v16h v; v8h h[2]; } u;
                const _Float16* p = &Ks[si * 16 + n][cf * 32 + ks];
                u.h[0] = *(const v8h*)(p);
                u.h[1] = *(const v8h*)(p + 16);
                s = wmma_f16(cf ? qa1 : qa0, u.v, s);
            }
            st[si] = s;
        }

        // ---- online softmax per owned row (row = j + 8*hi) ----
        #pragma unroll
        for (int j = 0; j < 8; ++j) {
            float v0 = st[0][j], v1 = st[1][j];
            float mx = fmaxf(v0, v1);
            #pragma unroll
            for (int off = 1; off < 16; off <<= 1)
                mx = fmaxf(mx, __shfl_xor(mx, off, 32));
            const float newm = fmaxf(m8[j], mx);
            const float corr = exp2f((m8[j] - newm) * L2E);
            const float p0 = exp2f((v0 - newm) * L2E);
            const float p1 = exp2f((v1 - newm) * L2E);
            float rs = p0 + p1;
            #pragma unroll
            for (int off = 1; off < 16; off <<= 1)
                rs += __shfl_xor(rs, off, 32);
            l8[j] = l8[j] * corr + rs;
            m8[j] = newm;
            #pragma unroll
            for (int nn = 0; nn < 4; ++nn) acc[nn][j] *= corr;
            const int row = j + 8 * hi;
            Pb[w][row][n]      = (_Float16)p0;
            Pb[w][row][16 + n] = (_Float16)p1;
        }

        // P (D layout) -> A fragment via LDS (per-wave DS ops are in order)
        union { v16h v; v8h h[2]; } pu;
        {
            const int kb = hi * 8;
            pu.h[0] = *(const v8h*)&Pb[w][n][kb];
            pu.h[1] = *(const v8h*)&Pb[w][n][kb + 16];
        }

        // ---- acc += P(16x32) x V^T(32x16), V from LDS ----
        #pragma unroll
        for (int nn = 0; nn < 4; ++nn) {
            union { v16h v; v8h h[2]; } u;
            const _Float16* p = &Vs[nn * 16 + n][ks];
            u.h[0] = *(const v8h*)(p);
            u.h[1] = *(const v8h*)(p + 16);
            acc[nn] = wmma_f16(pu.v, u.v, acc[nn]);
        }
        __syncthreads();   // protect staged tiles before next overwrite
    }

    // normalize and write h[b][t][c] f16
    const int b    = bh >> 3;
    const int head = bh & 7;
    #pragma unroll
    for (int j = 0; j < 8; ++j) {
        const float inv = 1.0f / l8[j];
        const int t = t0 + j + 8 * hi;
        #pragma unroll
        for (int nn = 0; nn < 4; ++nn) {
            const int c = head * HD + nn * 16 + n;
            HT[((size_t)b * TT + t) * CC + c] = (_Float16)(acc[nn][j] * inv);
        }
    }
}

// ---------------------------------------------------------------------------
// K4: output projection + bias + residual (f32 out).
// out[b,o,t] = x[b,o,t] + b_proj[o] + sum_c Wproj[o,c] * h[b,t,c]
// ---------------------------------------------------------------------------
__global__ void k_proj_residual(const float* __restrict__ x,
                                const float* __restrict__ b_proj,
                                const char* __restrict__ ws,
                                float* __restrict__ out) {
    const _Float16* WP = (const _Float16*)(ws + OFF_WP);
    const _Float16* HT = (const _Float16*)(ws + OFF_HT);

    const int gw = blockIdx.x * 8 + (threadIdx.x >> 5);
    const int tt = gw & 15;
    const int ct = (gw >> 4) & 31;
    const int b  = gw >> 9;
    const int o0 = ct * 16;
    const int t0 = tt * 64;

    v8f acc[4];
    #pragma unroll
    for (int i = 0; i < 4; ++i) acc[i] = (v8f)(0.0f);

    for (int kc = 0; kc < CC; kc += 32) {
        v16h a = load_frag_a(WP + (size_t)o0 * CC + kc, CC);
        #pragma unroll
        for (int nn = 0; nn < 4; ++nn) {
            v16h bf = load_frag_b(HT + ((size_t)b * TT + t0 + nn * 16) * CC + kc, CC);
            acc[nn] = wmma_f16(a, bf, acc[nn]);
        }
    }

    const int l  = threadIdx.x & 31;
    const int hi = l >> 4;
    const int n  = l & 15;
    #pragma unroll
    for (int nn = 0; nn < 4; ++nn) {
        #pragma unroll
        for (int j = 0; j < 8; ++j) {
            const int o = o0 + j + 8 * hi;
            const int t = t0 + nn * 16 + n;
            const size_t idx = ((size_t)b * CC + o) * TT + t;
            out[idx] = x[idx] + b_proj[o] + acc[nn][j];
        }
    }
}

// ---------------------------------------------------------------------------
extern "C" void kernel_launch(void* const* d_in, const int* in_sizes, int n_in,
                              void* d_out, int out_size, void* d_ws, size_t ws_size,
                              hipStream_t stream) {
    const float* x      = (const float*)d_in[0];
    const float* emb    = (const float*)d_in[1];  (void)emb;  // unused by reference
    const float* w_qkv  = (const float*)d_in[2];
    const float* b_qkv  = (const float*)d_in[3];
    const float* w_proj = (const float*)d_in[4];
    const float* b_proj = (const float*)d_in[5];
    float* out = (float*)d_out;
    char* ws   = (char*)d_ws;
    (void)in_sizes; (void)n_in; (void)out_size; (void)ws_size;

    // K0: weight f32 -> f16  (1,048,576 elems)
    k_prep_weights<<<4096, 256, 0, stream>>>(w_qkv, w_proj, ws);
    // K1: layernorm + transpose to [b][t][c] f16
    k_layernorm<<<BB * (TT / 64), 64, 0, stream>>>(x, ws);
    // K2: QKV GEMM (24576 wave-tiles / 8 waves per block)
    k_qkv_gemm<<<3072, 256, 0, stream>>>(b_qkv, ws);
    // K3: flash attention (128 heads x 16 t-tiles)
    k_attention<<<2048, 128, 0, stream>>>(ws);
    // K4: projection + residual (8192 wave-tiles / 8)
    k_proj_residual<<<1024, 256, 0, stream>>>(x, b_proj, ws, out);
}
